// PokerQNetwork_26817775796697
// MI455X (gfx1250) — matrix operand order repaired and checked by reading, not compiled, not run on hardware
//
#include <hip/hip_runtime.h>

typedef __attribute__((ext_vector_type(16))) __bf16 v16bf;
typedef __attribute__((ext_vector_type(8)))  __bf16 v8bf;
typedef __attribute__((ext_vector_type(8)))  float  v8f;

#define ROWS    64
#define THREADS 256
#define NWAVES  8
#define MTILES  (ROWS / 16)

// bf16 weight arena layout in d_ws (element offsets)
#define WOFF1 0         // fc1_w: 512*128
#define WOFF2 65536     // fc2_w: 256*512
#define WOFF3 196608    // fc3_w: 128*256
#define WOFF4 229376    // fc4_w: 3*128
#define WTOT  229760    // total bf16 elements

// ---------------------------------------------------------------------------
// fp32 -> bf16 weight conversion (runs once per launch; idempotent)
// ---------------------------------------------------------------------------
__global__ void cvt_w_bf16(const float* __restrict__ w, __bf16* __restrict__ o, int n) {
    const int i = blockIdx.x * 256 + threadIdx.x;
    if (i < n) o[i] = (__bf16)w[i];
}

// ---------------------------------------------------------------------------
// One FC layer:  D[ROWS][N] = act( A[ROWS][K] @ W[N][K]^T + bias )
// A: bf16 LDS row-major (stride K). Weights: bf16 (BW=true) or fp32 fallback.
// Each wave owns M-tile (wave % MTILES) and strides over N-tiles, so the
// nt-invariant A fragments (ds_load_b128 pairs) can be hoisted/reused.
// ---------------------------------------------------------------------------
template<bool BW>
__device__ __forceinline__ void gemm_tile_layer(
    const __bf16* __restrict__ A, int K,
    const __bf16* __restrict__ Wb, const float* __restrict__ W32,
    const float*  __restrict__ bias, int N,
    __bf16* __restrict__ D, bool relu,
    float* __restrict__ gout, int base_row)
{
    const int tid  = threadIdx.x;
    const int wave = tid >> 5;
    const int lane = tid & 31;
    const int half = lane >> 4;          // 16-lane half of the wave
    const int nloc = lane & 15;          // column within N-tile

    const int Ntiles = (N + 15) / 16;
    const int Kb     = K >> 5;           // K-blocks of 32

    const int mt  = wave % MTILES;       // fixed M-tile per wave
    const int nt0 = wave / MTILES;       // 0 or 1
    const int m   = mt * 16 + (lane & 15);
    const __bf16* arow = A + (size_t)m * K;   // + kb*32 + half*16 inside

    for (int nt = nt0; nt < Ntiles; nt += NWAVES / MTILES) {
        const int colr = nt * 16 + nloc;
        const int colc = (colr < N) ? colr : (N - 1);   // clamp (fc4 N=3)

        v8f acc = {};
        for (int kb = 0; kb < Kb; ++kb) {
            // A 16x32 bf16 layout: lane half selects K sub-ranges; per lane the
            // fragment is two contiguous 8xbf16 runs -> two ds_load_b128.
            const __bf16* ap = arow + kb * 32;
            const v8bf alo = *(const v8bf*)(ap + half * 16);
            const v8bf ahi = *(const v8bf*)(ap + half * 16 + 8);
            const v16bf a = __builtin_shufflevector(alo, ahi,
                0,1,2,3,4,5,6,7,8,9,10,11,12,13,14,15);

            // B 32x16 bf16 layout: lane column colc, K = half*16 + 0..15 ->
            // one contiguous 32-byte run in [N][K] row-major.
            v16bf b;
            if (BW) {
                const __bf16* wb = Wb + (size_t)colc * K + kb * 32 + half * 16;
                const v8bf blo = *(const v8bf*)(wb);
                const v8bf bhi = *(const v8bf*)(wb + 8);
                b = __builtin_shufflevector(blo, bhi,
                    0,1,2,3,4,5,6,7,8,9,10,11,12,13,14,15);
            } else {
                const float* wp = W32 + (size_t)colc * K + kb * 32;
#pragma unroll
                for (int j = 0; j < 8; ++j) {
                    const int kw = half * 16 + 2 * j;
                    b[2 * j]     = (__bf16)wp[kw];
                    b[2 * j + 1] = (__bf16)wp[kw + 1];
                }
            }
            acc = __builtin_amdgcn_wmma_f32_16x16x32_bf16(
                false, a, false, b, (short)0, acc, false, false);
        }

        const float bv = bias[colc];
        if (gout == nullptr) {
            // C/D layout: VGPR rr -> row mt*16 + rr + 8*half, col = colr.
#pragma unroll
            for (int rr = 0; rr < 8; ++rr) {
                const int row = mt * 16 + rr + half * 8;
                float v = acc[rr] + bv;
                if (relu) v = fmaxf(v, 0.0f);
                D[(size_t)row * N + colr] = (__bf16)v;
            }
        } else if (colr < N) {
#pragma unroll
            for (int rr = 0; rr < 8; ++rr) {
                const int row = mt * 16 + rr + half * 8;
                gout[(size_t)(base_row + row) * N + colr] = acc[rr] + bv;
            }
        }
    }
}

// ---------------------------------------------------------------------------
// Fused network: front-end (VALU, 4 threads/row) -> comb[128] bf16 in LDS ->
// WMMA FC stack (bf16 A/B, f32 accumulate). 256 threads, 64 rows per block.
// ---------------------------------------------------------------------------
template<bool BW>
__global__ __launch_bounds__(THREADS)
void poker_qnet_fused(
    const float* __restrict__ card_input, const float* __restrict__ state_input,
    const float* __restrict__ emb,
    const float* __restrict__ hole_w, const float* __restrict__ hole_b,
    const float* __restrict__ comm_w, const float* __restrict__ comm_b,
    const float* __restrict__ in_proj_w, const float* __restrict__ in_proj_b,
    const float* __restrict__ out_proj_w, const float* __restrict__ out_proj_b,
    const float* __restrict__ state_w, const float* __restrict__ state_b,
    const float* __restrict__ ln_g, const float* __restrict__ ln_b,
    const float* __restrict__ fc1_w, const float* __restrict__ fc1_b,
    const float* __restrict__ fc2_w, const float* __restrict__ fc2_b,
    const float* __restrict__ fc3_w, const float* __restrict__ fc3_b,
    const float* __restrict__ fc4_w, const float* __restrict__ fc4_b,
    const __bf16* __restrict__ wbf,
    float* __restrict__ out)
{
    __shared__ __align__(16) unsigned char sX[65536];
    __shared__ __align__(16) unsigned char sY[65536];

    // Arena X: comb bf16 [64][128] @0 (16KB); comb f32 [64][128] @32768 (32KB).
    // Later: h2 bf16 [64][256] @0 (32KB).
    __bf16* combA = (__bf16*)sX;
    float*  cf32  = (float*)(sX + 32768);
    // Arena Y front-end scratch; later h1 bf16 [64][512] (64KB) / h3 [64][128].
    float* sCE = (float*)sY;            // [64][7][16]  f32  (28672 B)
    float* xb  = (float*)(sY + 28672);  // [64][2][32]  f32  (16384 B)
    float* ob  = (float*)(sY + 45056);  // [64][2][32]  f32  (16384 B)
    float* red = (float*)(sY + 61440);  // [64][4][2]   f32  ( 2048 B)

    const int tid = threadIdx.x;
    const int r   = tid >> 2;           // row within block [0,64)
    const int p   = tid & 3;            // 4-way split within a row
    const int gr  = blockIdx.x * ROWS + r;

    // ---- Step A: card ids + embedding gather (p covers 4 emb columns) ----
    {
        int id[7];
#pragma unroll
        for (int c = 0; c < 7; ++c) {
            const float v0 = card_input[(size_t)gr * 14 + c];
            const float v1 = card_input[(size_t)gr * 14 + 7 + c];
            const int suit = (int)(v0 * 3.0f);
            const int rank = (int)(v1 * 12.0f);
            id[c] = ((v0 > 0.0f) || (v1 > 0.0f)) ? (rank * 4 + suit + 1) : 0;
        }
#pragma unroll
        for (int c = 0; c < 7; ++c) {
            const float* e = emb + id[c] * 16 + p * 4;
            float* dst = sCE + (r * 7 + c) * 16 + p * 4;
            dst[0] = e[0]; dst[1] = e[1]; dst[2] = e[2]; dst[3] = e[3];
        }
    }
    __syncthreads();

    // ---- Step B: hole_feat + comm conv/mean (p covers 8 out-channels) ----
    {
        const float* ce0 = sCE + (r * 7 + 0) * 16;
        const float* ce1 = sCE + (r * 7 + 1) * 16;
        for (int oo = 0; oo < 8; ++oo) {
            const int o = p * 8 + oo;
            float hf = hole_b[o];
            const float* hw = hole_w + o * 32;           // [o][i][k]
            for (int i = 0; i < 16; ++i)
                hf += ce0[i] * hw[i * 2 + 0] + ce1[i] * hw[i * 2 + 1];

            float cfsum = 5.0f * comm_b[o];
            const float* cw = comm_w + o * 48;           // [o][i][t]
            for (int pos = 0; pos < 5; ++pos) {
                for (int t = 0; t < 3; ++t) {
                    const int src = pos + t - 1;
                    if (src < 0 || src >= 5) continue;
                    const float* ce = sCE + (r * 7 + 2 + src) * 16;
                    for (int i = 0; i < 16; ++i)
                        cfsum += ce[i] * cw[i * 3 + t];
                }
            }
            xb[(r * 2 + 0) * 32 + o] = hf;
            xb[(r * 2 + 1) * 32 + o] = cfsum * 0.2f;     // mean over 5 positions
            cf32[r * 128 + o] = hf;                      // comb[0:32) = hole_feat
        }
    }
    __syncthreads();

    // ---- Step C: self-attention, head h = p (seq len 2, head_dim 8) ----
    {
        const int h = p;
        float q[2][8], kk[2][8], vv[2][8];
        for (int s = 0; s < 2; ++s) {
            const float* x = xb + (r * 2 + s) * 32;
            for (int d = 0; d < 8; ++d) {
                const int jq = 8 * h + d, jk = 32 + 8 * h + d, jv = 64 + 8 * h + d;
                float aq = in_proj_b[jq], ak = in_proj_b[jk], av = in_proj_b[jv];
                const float* wq = in_proj_w + jq * 32;
                const float* wk = in_proj_w + jk * 32;
                const float* wv = in_proj_w + jv * 32;
                for (int i = 0; i < 32; ++i) {
                    const float xi = x[i];
                    aq += xi * wq[i]; ak += xi * wk[i]; av += xi * wv[i];
                }
                q[s][d] = aq; kk[s][d] = ak; vv[s][d] = av;
            }
        }
        const float scale = 0.35355339059f;              // 1/sqrt(8)
        for (int s = 0; s < 2; ++s) {
            float s0 = 0.0f, s1 = 0.0f;
            for (int d = 0; d < 8; ++d) { s0 += q[s][d] * kk[0][d]; s1 += q[s][d] * kk[1][d]; }
            s0 *= scale; s1 *= scale;
            const float m  = fmaxf(s0, s1);
            const float e0 = __expf(s0 - m), e1 = __expf(s1 - m);
            const float inv = 1.0f / (e0 + e1);
            const float a0 = e0 * inv, a1 = e1 * inv;
            for (int d = 0; d < 8; ++d)
                ob[(r * 2 + s) * 32 + 8 * h + d] = a0 * vv[0][d] + a1 * vv[1][d];
        }
    }
    __syncthreads();

    // ---- Step D/E: out_proj + token-mean; state MLP (relu) ----
    {
        const float* o0 = ob + (r * 2 + 0) * 32;
        const float* o1 = ob + (r * 2 + 1) * 32;
        for (int oo = 0; oo < 8; ++oo) {
            const int o = p * 8 + oo;
            const float* w = out_proj_w + o * 32;
            float a0 = 0.0f, a1 = 0.0f;
            for (int i = 0; i < 32; ++i) { a0 += o0[i] * w[i]; a1 += o1[i] * w[i]; }
            cf32[r * 128 + 32 + o] = out_proj_b[o] + 0.5f * (a0 + a1);
        }
        const float* st = state_input + (size_t)gr * 10;
        for (int jj = 0; jj < 16; ++jj) {
            const int j = p * 16 + jj;
            const float* w = state_w + j * 10;
            float sf = state_b[j];
            for (int i = 0; i < 10; ++i) sf += st[i] * w[i];
            cf32[r * 128 + 64 + j] = fmaxf(sf, 0.0f);
        }
    }
    __syncthreads();

    // ---- Step F: LayerNorm(128) -> bf16 A-matrix ----
    {
        float s = 0.0f, ss = 0.0f;
        const float* c = cf32 + r * 128 + p * 32;
        for (int i = 0; i < 32; ++i) { const float v = c[i]; s += v; ss += v * v; }
        red[(r * 4 + p) * 2 + 0] = s;
        red[(r * 4 + p) * 2 + 1] = ss;
    }
    __syncthreads();
    {
        float s = 0.0f, ss = 0.0f;
        for (int q2 = 0; q2 < 4; ++q2) {
            s  += red[(r * 4 + q2) * 2 + 0];
            ss += red[(r * 4 + q2) * 2 + 1];
        }
        const float mean = s * (1.0f / 128.0f);
        const float var  = ss * (1.0f / 128.0f) - mean * mean;
        const float rstd = rsqrtf(var + 1e-5f);
        const float* c = cf32 + r * 128;
        for (int ii = 0; ii < 32; ++ii) {
            const int idx = p * 32 + ii;
            const float v = (c[idx] - mean) * rstd * ln_g[idx] + ln_b[idx];
            combA[r * 128 + idx] = (__bf16)v;
        }
    }
    __syncthreads();

    // ---- WMMA FC stack ----
    const __bf16* w1 = BW ? (wbf + WOFF1) : nullptr;
    const __bf16* w2 = BW ? (wbf + WOFF2) : nullptr;
    const __bf16* w3 = BW ? (wbf + WOFF3) : nullptr;
    const __bf16* w4 = BW ? (wbf + WOFF4) : nullptr;

    __bf16* h1 = (__bf16*)sY;                       // [64][512] (64KB, scratch dead)
    gemm_tile_layer<BW>(combA, 128, w1, fc1_w, fc1_b, 512, h1, true, nullptr, 0);
    __syncthreads();
    __bf16* h2 = (__bf16*)sX;                       // [64][256] (32KB)
    gemm_tile_layer<BW>(h1, 512, w2, fc2_w, fc2_b, 256, h2, true, nullptr, 0);
    __syncthreads();
    __bf16* h3 = (__bf16*)sY;                       // [64][128] (16KB)
    gemm_tile_layer<BW>(h2, 256, w3, fc3_w, fc3_b, 128, h3, true, nullptr, 0);
    __syncthreads();
    gemm_tile_layer<BW>(h3, 128, w4, fc4_w, fc4_b, 3, nullptr, false, out, blockIdx.x * ROWS);
}

extern "C" void kernel_launch(void* const* d_in, const int* in_sizes, int n_in,
                              void* d_out, int out_size, void* d_ws, size_t ws_size,
                              hipStream_t stream) {
    const float* card_input = (const float*)d_in[0];
    const float* state_input= (const float*)d_in[1];
    const float* emb        = (const float*)d_in[2];
    const float* hole_w     = (const float*)d_in[3];
    const float* hole_b     = (const float*)d_in[4];
    const float* comm_w     = (const float*)d_in[5];
    const float* comm_b     = (const float*)d_in[6];
    const float* in_proj_w  = (const float*)d_in[7];
    const float* in_proj_b  = (const float*)d_in[8];
    const float* out_proj_w = (const float*)d_in[9];
    const float* out_proj_b = (const float*)d_in[10];
    const float* state_w    = (const float*)d_in[11];
    const float* state_b    = (const float*)d_in[12];
    const float* ln_g       = (const float*)d_in[13];
    const float* ln_b       = (const float*)d_in[14];
    const float* fc1_w      = (const float*)d_in[15];
    const float* fc1_b      = (const float*)d_in[16];
    const float* fc2_w      = (const float*)d_in[17];
    const float* fc2_b      = (const float*)d_in[18];
    const float* fc3_w      = (const float*)d_in[19];
    const float* fc3_b      = (const float*)d_in[20];
    const float* fc4_w      = (const float*)d_in[21];
    const float* fc4_b      = (const float*)d_in[22];

    const int B = in_sizes[0] / 14;                 // card_input is [B,2,7]
    const int grid = (B + ROWS - 1) / ROWS;

    const bool use_bw = (ws_size >= (size_t)WTOT * sizeof(__bf16));
    __bf16* wbf = (__bf16*)d_ws;

    if (use_bw) {
        // One-time per launch (idempotent, deterministic) fp32->bf16 weights.
        cvt_w_bf16<<<(65536  + 255) / 256, 256, 0, stream>>>(fc1_w, wbf + WOFF1, 65536);
        cvt_w_bf16<<<(131072 + 255) / 256, 256, 0, stream>>>(fc2_w, wbf + WOFF2, 131072);
        cvt_w_bf16<<<(32768  + 255) / 256, 256, 0, stream>>>(fc3_w, wbf + WOFF3, 32768);
        cvt_w_bf16<<<(384    + 255) / 256, 256, 0, stream>>>(fc4_w, wbf + WOFF4, 384);

        poker_qnet_fused<true><<<grid, THREADS, 0, stream>>>(
            card_input, state_input, emb, hole_w, hole_b, comm_w, comm_b,
            in_proj_w, in_proj_b, out_proj_w, out_proj_b, state_w, state_b,
            ln_g, ln_b, fc1_w, fc1_b, fc2_w, fc2_b, fc3_w, fc3_b, fc4_w, fc4_b,
            wbf, (float*)d_out);
    } else {
        poker_qnet_fused<false><<<grid, THREADS, 0, stream>>>(
            card_input, state_input, emb, hole_w, hole_b, comm_w, comm_b,
            in_proj_w, in_proj_b, out_proj_w, out_proj_b, state_w, state_b,
            ln_g, ln_b, fc1_w, fc1_b, fc2_w, fc2_b, fc3_w, fc3_b, fc4_w, fc4_b,
            nullptr, (float*)d_out);
    }
}
